// MultiHeadSelfAttention_4063039062684
// MI455X (gfx1250) — matrix operand order repaired and checked by reading, not compiled, not run on hardware
//
#include <hip/hip_runtime.h>

typedef __attribute__((ext_vector_type(16))) __bf16 v16bf;
typedef __attribute__((ext_vector_type(8)))  __bf16 v8bf;
typedef __attribute__((ext_vector_type(8)))  float  v8f;

#define BB 4
#define TT 2048
#define DD 768
#define HH 12
#define HD 64
#define MROWS (BB * TT)      /* 8192 */
#define NQKV  (3 * DD)       /* 2304 */

__device__ __forceinline__ v16bf cat16(v8bf lo, v8bf hi) {
  v16bf r;
#pragma unroll
  for (int i = 0; i < 8; ++i) { r[i] = lo[i]; r[i + 8] = hi[i]; }
  return r;
}

__device__ __forceinline__ v8f wmma_bf16(v16bf a, v16bf b, v8f c) {
  return __builtin_amdgcn_wmma_f32_16x16x32_bf16(false, a, false, b, (short)0, c,
                                                 false, false);
}

// Async global->LDS copy of 16B (CDNA5 GLOBAL_LOAD_ASYNC_TO_LDS_B128, GV mode).
// INST_OFFSET applies to both the LDS and the memory address (ISA 15.18.3).
__device__ __forceinline__ void async_copy16(unsigned lds_off,
                                             const void* gaddr) {
  asm volatile("global_load_async_to_lds_b128 %0, %1, off"
               :
               : "v"(lds_off), "v"(gaddr)
               : "memory");
}
__device__ __forceinline__ void async_copy16_off16(unsigned lds_off,
                                                   const void* gaddr) {
  asm volatile("global_load_async_to_lds_b128 %0, %1, off offset:16"
               :
               : "v"(lds_off), "v"(gaddr)
               : "memory");
}
__device__ __forceinline__ void wait_async0() {
  asm volatile("s_wait_asynccnt 0" ::: "memory");
}

// Lane-XOR exchange via DS_SWIZZLE_B32 (group-of-32: and=0x1f, or=0, xor=MASK).
// One DS op per step; masks 1/2/4/8 stay inside each 16-lane row group.
template <int MASK>
__device__ __forceinline__ float swz_xor(float v) {
  return __int_as_float(
      __builtin_amdgcn_ds_swizzle(__float_as_int(v), (MASK << 10) | 0x1f));
}
__device__ __forceinline__ float rowmax16(float t) {
  t = fmaxf(t, swz_xor<1>(t));
  t = fmaxf(t, swz_xor<2>(t));
  t = fmaxf(t, swz_xor<4>(t));
  t = fmaxf(t, swz_xor<8>(t));
  return t;
}
__device__ __forceinline__ float rowsum16(float t) {
  t += swz_xor<1>(t);
  t += swz_xor<2>(t);
  t += swz_xor<4>(t);
  t += swz_xor<8>(t);
  return t;
}

// Online-softmax update for one 16x32 score block.  MASKED=false is the fast
// path for tiles entirely below the causal diagonal (no compares/cndmasks).
template <bool MASKED>
__device__ __forceinline__ void softmax_step(const v8f& S0, const v8f& S1,
                                             float* __restrict__ m,
                                             float* __restrict__ l,
                                             v8f* __restrict__ o,
                                             __bf16 (*__restrict__ ldsPw)[32],
                                             int q0, int kb0, int hl, int l16) {
#pragma unroll
  for (int i = 0; i < 8; ++i) {
    float s0 = S0[i], s1 = S1[i];
    bool m0 = false, m1 = false;
    if constexpr (MASKED) {
      const int rowq = q0 + i + hl * 8;
      m0 = (kb0 + l16) > rowq;
      m1 = (kb0 + 16 + l16) > rowq;
      s0 = m0 ? -3.0e38f : s0;
      s1 = m1 ? -3.0e38f : s1;
    }
    const float t = rowmax16(fmaxf(s0, s1));
    const float mnew = fmaxf(m[i], t);
    const float alpha = __expf(m[i] - mnew);
    float p0 = __expf(s0 - mnew);
    float p1 = __expf(s1 - mnew);
    if constexpr (MASKED) {
      p0 = m0 ? 0.0f : p0;
      p1 = m1 ? 0.0f : p1;
    }
    const float rs = rowsum16(p0 + p1);
    l[i] = l[i] * alpha + rs;
    m[i] = mnew;
#pragma unroll
    for (int nt = 0; nt < 4; ++nt) o[nt][i] *= alpha;
    const int prow = i + hl * 8;
    ldsPw[prow][l16]      = (__bf16)p0;
    ldsPw[prow][16 + l16] = (__bf16)p1;
  }
}

// ---------------------------------------------------------------- convert
// 8 floats per thread -> v_cvt_pk_bf16_f32 + b128 store.
__global__ void mhsa_cvt_f32_bf16(const float* __restrict__ in,
                                  __bf16* __restrict__ out, int n) {
  int i = (blockIdx.x * blockDim.x + threadIdx.x) * 8;
  if (i + 8 <= n) {
    float4 a = *(const float4*)(in + i);
    float4 b = *(const float4*)(in + i + 4);
    v8bf r;
    r[0] = (__bf16)a.x; r[1] = (__bf16)a.y; r[2] = (__bf16)a.z; r[3] = (__bf16)a.w;
    r[4] = (__bf16)b.x; r[5] = (__bf16)b.y; r[6] = (__bf16)b.z; r[7] = (__bf16)b.w;
    *(v8bf*)(out + i) = r;
  } else {
    for (; i < n; ++i) out[i] = (__bf16)in[i];
  }
}

// ---------------------------------------------------------------- GEMM
// C[M][N] = A[M][K] (bf16) x Bm[K][N] (bf16) + bias, f32 accumulate.
// WG = 256 threads (8 waves), tile 128x128, K-step 32. N,K compile-time so the
// epilogue is base-address + immediate-offset stores.
template <int N, int K, bool F32OUT>
__global__ __launch_bounds__(256) void mhsa_gemm_wmma(
    const __bf16* __restrict__ A, const __bf16* __restrict__ Bm,
    const float* __restrict__ bias, void* __restrict__ Cout) {
  __shared__ __bf16 ldsA[128][32];   // [row][k]
  __shared__ __bf16 ldsBt[128][32];  // [col][k]  (transposed stage)

  const int tid = threadIdx.x;
  const int lane = tid & 31, w = tid >> 5;
  const int l16 = lane & 15, hl = lane >> 4;
  const int bm = blockIdx.y * 128, bn = blockIdx.x * 128;
  const int waveM = (w & 3) * 32, waveN = (w >> 2) * 64;

  v8f acc[2][4];
#pragma unroll
  for (int mt = 0; mt < 2; ++mt)
#pragma unroll
    for (int nt = 0; nt < 4; ++nt) acc[mt][nt] = (v8f){};

  // Per-thread staging pointers, advanced by constant strides per K step.
  const int arow = tid >> 1, acol = (tid & 1) * 16;
  const unsigned aldsoff = (unsigned)(size_t)&ldsA[arow][acol];
  const int bk = tid & 31, bnb = (tid >> 5) * 16;
  const __bf16* asrc = A + (size_t)(bm + arow) * K + acol;
  const __bf16* bsrc = Bm + (size_t)bk * N + bn + bnb;

  for (int k0 = 0; k0 < K; k0 += 32) {
    // stage A tile via CDNA5 async global->LDS DMA (2 x 16B per thread)
    async_copy16(aldsoff, asrc);
    async_copy16_off16(aldsoff, asrc);
    {  // stage B tile transposed: each thread one k, 16 n's
      v8bf lo = *(const v8bf*)bsrc;
      v8bf hi = *(const v8bf*)(bsrc + 8);
#pragma unroll
      for (int i = 0; i < 8; ++i) {
        ldsBt[bnb + i][bk]     = lo[i];
        ldsBt[bnb + 8 + i][bk] = hi[i];
      }
    }
    asrc += 32;
    bsrc += (size_t)32 * N;
    wait_async0();   // ASYNCcnt not covered by __syncthreads
    __syncthreads();

    v16bf aF[2], bF[4];
    const int b0 = hl * 8;
#pragma unroll
    for (int mt = 0; mt < 2; ++mt) {
      const int row = waveM + mt * 16 + l16;
      aF[mt] = cat16(*(const v8bf*)&ldsA[row][b0],
                     *(const v8bf*)&ldsA[row][b0 + 16]);
    }
    const int kb = hl * 8;
#pragma unroll
    for (int nt = 0; nt < 4; ++nt) {
      const int col = waveN + nt * 16 + l16;
      bF[nt] = cat16(*(const v8bf*)&ldsBt[col][kb],
                     *(const v8bf*)&ldsBt[col][kb + 16]);
    }
#pragma unroll
    for (int mt = 0; mt < 2; ++mt)
#pragma unroll
      for (int nt = 0; nt < 4; ++nt)
        acc[mt][nt] = wmma_bf16(aF[mt], bF[nt], acc[mt][nt]);
    __syncthreads();
  }

  // epilogue: C layout -> row = i + 8*(lane>=16), col = lane%16
#pragma unroll
  for (int mt = 0; mt < 2; ++mt)
#pragma unroll
    for (int nt = 0; nt < 4; ++nt) {
      const int colg = bn + waveN + nt * 16 + l16;
      const int rowg0 = bm + waveM + mt * 16 + hl * 8;
      const float bv = bias[colg];
      if constexpr (F32OUT) {
        float* p = (float*)Cout + (size_t)rowg0 * N + colg;
#pragma unroll
        for (int i = 0; i < 8; ++i) p[i * N] = acc[mt][nt][i] + bv;
      } else {
        __bf16* p = (__bf16*)Cout + (size_t)rowg0 * N + colg;
#pragma unroll
        for (int i = 0; i < 8; ++i) p[i * N] = (__bf16)(acc[mt][nt][i] + bv);
      }
    }
}

// ---------------------------------------------------------------- attention
// grid (T/128, B*H); 256 threads = 8 waves, wave w owns queries q0..q0+15.
// K tile async-staged to LDS (shared by all 8 waves); V tile staged transposed.
__global__ __launch_bounds__(256) void mhsa_flash_attn_wmma(
    const __bf16* __restrict__ QKV, __bf16* __restrict__ O) {
  __shared__ __bf16 ldsK[32][64];       // [key][dim], async staged
  __shared__ __bf16 ldsVt[64][32];      // [dim][key]
  __shared__ __bf16 ldsP[8][16][32];    // per-wave P re-layout buffer

  const int tid = threadIdx.x, lane = tid & 31, w = tid >> 5;
  const int l16 = lane & 15, hl = lane >> 4;
  const int bh = blockIdx.y, b = bh / HH, h = bh % HH;
  const int qt = blockIdx.x;
  const int q0 = qt * 128 + w * 16;
  const __bf16* base = QKV + (size_t)b * TT * NQKV;

  // Q fragments (K-dim 64 = two K=32 frags), scale 1/sqrt(64) folded in.
  v16bf qF[2];
  {
    const int qrow = q0 + l16, b0 = hl * 8;
    const __bf16* qptr = base + (size_t)qrow * NQKV + h * HD;
#pragma unroll
    for (int f = 0; f < 2; ++f) {
      v8bf lo = *(const v8bf*)(qptr + f * 32 + b0);
      v8bf hi = *(const v8bf*)(qptr + f * 32 + b0 + 16);
#pragma unroll
      for (int i = 0; i < 8; ++i) {
        qF[f][i]     = (__bf16)((float)lo[i] * 0.125f);
        qF[f][i + 8] = (__bf16)((float)hi[i] * 0.125f);
      }
    }
  }

  float m[8], l[8];
  v8f o[4];
#pragma unroll
  for (int i = 0; i < 8; ++i) { m[i] = -3.0e38f; l[i] = 0.0f; }
#pragma unroll
  for (int nt = 0; nt < 4; ++nt) o[nt] = (v8f){};

  // Per-thread staging pointers, advanced by 32 rows per tile.
  const int skey = tid >> 3, sdb = (tid & 7) * 8;   // K stage: 16B/thread
  const unsigned kldsoff = (unsigned)(size_t)&ldsK[skey][sdb];
  const int vkey = tid & 31, vdb = (tid >> 5) * 8;  // V stage
  const __bf16* ksrc = base + (size_t)skey * NQKV + DD + h * HD + sdb;
  const __bf16* vsrc = base + (size_t)vkey * NQKV + 2 * DD + h * HD + vdb;

  const int nkt = qt * 4 + 4;  // key tiles needed by this WG (causal)
  for (int kt = 0; kt < nkt; ++kt) {
    const int kb0 = kt * 32;
    // async stage of K tile (32 keys x 64 dims) -> LDS, no transpose needed
    async_copy16(kldsoff, ksrc);
    // unconditional speculative prefetch of next tile (stays inside ws)
    __builtin_prefetch(ksrc + 32 * NQKV, 0, 1);
    {  // cooperative stage of V^T tile (32 keys x 64 dims)
      v8bf v = *(const v8bf*)vsrc;
      __builtin_prefetch(vsrc + 32 * NQKV, 0, 1);
#pragma unroll
      for (int i = 0; i < 8; ++i) ldsVt[vdb + i][vkey] = v[i];
    }
    ksrc += (size_t)32 * NQKV;
    vsrc += (size_t)32 * NQKV;
    wait_async0();   // ASYNCcnt not covered by __syncthreads
    __syncthreads();

    if (kb0 <= q0 + 15) {  // tile not entirely above the causal diagonal
      // ---- S = Q * K^T : preload all K^T B-frags, then chain WMMAs
      v16bf kF[2][2];
#pragma unroll
      for (int s = 0; s < 2; ++s) {
        const int krow = s * 16 + l16;
#pragma unroll
        for (int f = 0; f < 2; ++f) {
          const __bf16* kp = &ldsK[krow][f * 32 + hl * 8];
          kF[s][f] = cat16(*(const v8bf*)kp, *(const v8bf*)(kp + 16));
        }
      }
      v8f S[2];
#pragma unroll
      for (int s = 0; s < 2; ++s) {
        v8f c = (v8f){};
        c = wmma_bf16(qF[0], kF[s][0], c);
        c = wmma_bf16(qF[1], kF[s][1], c);
        S[s] = c;
      }
      // ---- online softmax; mask path only on the diagonal tile
      if (kb0 + 31 <= q0)
        softmax_step<false>(S[0], S[1], m, l, o, ldsP[w], q0, kb0, hl, l16);
      else
        softmax_step<true>(S[0], S[1], m, l, o, ldsP[w], q0, kb0, hl, l16);

      // wave-internal LDS RAW: DS ops are in-order per wave; block reordering.
      __builtin_amdgcn_wave_barrier();
      asm volatile("s_wait_dscnt 0" ::: "memory");

      // ---- P A-frag (16x32 over this 32-key tile)
      const int b0 = hl * 8;
      v16bf pF = cat16(*(const v8bf*)&ldsP[w][l16][b0],
                       *(const v8bf*)&ldsP[w][l16][b0 + 16]);
      // ---- O += P * V : preload all V B-frags, then chain WMMAs
      v16bf vF[4];
      const int kb = hl * 8;
#pragma unroll
      for (int nt = 0; nt < 4; ++nt) {
        const int dim = nt * 16 + l16;
        vF[nt] = cat16(*(const v8bf*)&ldsVt[dim][kb],
                       *(const v8bf*)&ldsVt[dim][kb + 16]);
      }
#pragma unroll
      for (int nt = 0; nt < 4; ++nt) o[nt] = wmma_bf16(pF, vF[nt], o[nt]);
    }
    __syncthreads();
  }

  // ---- epilogue: O / l, merged-head layout [B*T][D]
  __bf16* obase = O + (size_t)b * TT * DD + h * HD + (size_t)q0 * DD;
#pragma unroll
  for (int nt = 0; nt < 4; ++nt) {
    const int dim = nt * 16 + l16;
    __bf16* p = obase + (hl * 8) * DD + dim;
#pragma unroll
    for (int i = 0; i < 8; ++i) p[i * DD] = (__bf16)(o[nt][i] / l[i]);
  }
}

// ---------------------------------------------------------------- launch
extern "C" void kernel_launch(void* const* d_in, const int* in_sizes, int n_in,
                              void* d_out, int out_size, void* d_ws,
                              size_t ws_size, hipStream_t stream) {
  (void)in_sizes; (void)n_in; (void)out_size; (void)ws_size;
  const float* x     = (const float*)d_in[0];
  // d_in[1] = tril mask: causal masking is computed analytically, not read.
  const float* W_qkv = (const float*)d_in[2];
  const float* b_qkv = (const float*)d_in[3];
  const float* W_out = (const float*)d_in[4];
  const float* b_out = (const float*)d_in[5];
  float* out = (float*)d_out;

  char* ws = (char*)d_ws;
  size_t off = 0;
  auto take = [&](size_t bytes) {
    char* p = ws + off;
    off = (off + bytes + 255) & ~(size_t)255;
    return p;
  };
  __bf16* Xbf    = (__bf16*)take((size_t)MROWS * DD * 2);
  __bf16* Wqkvbf = (__bf16*)take((size_t)DD * NQKV * 2);
  __bf16* Woutbf = (__bf16*)take((size_t)DD * DD * 2);
  __bf16* QKVbf  = (__bf16*)take((size_t)MROWS * NQKV * 2);
  __bf16* Obf    = (__bf16*)take((size_t)MROWS * DD * 2);

  // 1) fp32 -> bf16 conversions (8 elts/thread)
  {
    int n = MROWS * DD;
    mhsa_cvt_f32_bf16<<<(n / 8 + 255) / 256, 256, 0, stream>>>(x, Xbf, n);
    n = DD * NQKV;
    mhsa_cvt_f32_bf16<<<(n / 8 + 255) / 256, 256, 0, stream>>>(W_qkv, Wqkvbf, n);
    n = DD * DD;
    mhsa_cvt_f32_bf16<<<(n / 8 + 255) / 256, 256, 0, stream>>>(W_out, Woutbf, n);
  }
  // 2) QKV = X * W_qkv + b_qkv  (bf16 out)
  mhsa_gemm_wmma<NQKV, DD, false>
      <<<dim3(NQKV / 128, MROWS / 128), 256, 0, stream>>>(Xbf, Wqkvbf, b_qkv,
                                                          QKVbf);
  // 3) flash attention per (b,h)
  mhsa_flash_attn_wmma<<<dim3(TT / 128, BB * HH), 256, 0, stream>>>(QKVbf, Obf);
  // 4) out = O * W_out + b_out  (f32 out)
  mhsa_gemm_wmma<DD, DD, true>
      <<<dim3(DD / 128, MROWS / 128), 256, 0, stream>>>(Obf, Woutbf, b_out,
                                                        out);
}
